// SelfAttention_87256555585999
// MI455X (gfx1250) — matrix operand order repaired and checked by reading, not compiled, not run on hardware
//
#include <hip/hip_runtime.h>
#include <hip/hip_bf16.h>

// ---------------------------------------------------------------------------
// LUKE-style word/entity self-attention for MI455X (gfx1250, wave32, WMMA).
//   Stage 1: 12 projections  Y = X @ W^T + b  via v_wmma_f32_16x16x32_bf16
//   Stage 2: flash-style attention over the 2304-key axis with block-dependent
//            Q/K projections; K tiles staged via the Tensor Data Mover
//            (tensor_load_to_lds + s_wait_tensorcnt), online softmax, P*V WMMA.
// ---------------------------------------------------------------------------

#define NH    12
#define HD    64
#define HDIM  768
#define BATCH 4
#define LW    2048
#define LE    256
#define LSUM  (LW + LE)

typedef __attribute__((ext_vector_type(16))) __bf16   v16bf;
typedef __attribute__((ext_vector_type(8)))  __bf16   v8bf;
typedef __attribute__((ext_vector_type(8)))  float    v8f;
typedef __attribute__((ext_vector_type(4)))  unsigned u32x4;
typedef __attribute__((ext_vector_type(8)))  unsigned u32x8;

static __device__ inline v16bf cat8(v8bf lo, v8bf hi) {
  return __builtin_shufflevector(lo, hi, 0,1,2,3,4,5,6,7,8,9,10,11,12,13,14,15);
}

static __device__ inline v8f wmma_bf16(v16bf a, v16bf b, v8f c) {
  return __builtin_amdgcn_wmma_f32_16x16x32_bf16(false, a, false, b,
                                                 (short)0, c, false, false);
}

// LDS byte offset of a __shared__ object (TDM descriptors want the LDS-space
// address, not the generic-aperture address).
static __device__ inline unsigned lds_offset(const void* p) {
  return (unsigned)(unsigned long long)
      (__attribute__((address_space(3))) const void*)p;
}

// Issue a TDM 2D tile load: rows x cols bf16, row stride = cols elements,
// dense tile, no padding/iteration/multicast. Wave-uniform operands.
static __device__ inline void tdm_load_tile_bf16(unsigned lds_byte_addr,
                                                 const void* gsrc,
                                                 unsigned rows, unsigned cols) {
  unsigned long long ga = (unsigned long long)gsrc;
  u32x4 g0;
  g0[0] = 1u;                                   // count = 1 valid descriptor
  g0[1] = lds_byte_addr;                        // lds_addr
  g0[2] = (unsigned)ga;                         // global_addr[31:0]
  g0[3] = (unsigned)(ga >> 32) | 0x80000000u;   // global_addr[56:32] | type=2
  u32x8 g1;
  g1[0] = 1u << 16;                             // data_size = 1 (2 bytes)
  g1[1] = cols << 16;                           // tensor_dim0[15:0]  @ [63:48]
  g1[2] = rows << 16;                           // tensor_dim1[15:0]  @ [95:80]
  g1[3] = cols << 16;                           // tile_dim0          @ [127:112]
  g1[4] = rows;                                 // tile_dim1          @ [143:128]
  g1[5] = cols;                                 // tensor_dim0_stride @ [191:160]
  g1[6] = 0u;
  g1[7] = 0u;
  asm volatile("tensor_load_to_lds %0, %1" :: "s"(g0), "s"(g1) : "memory");
}

// ---------------------------------------------------------------------------
// Projection GEMM: dst[b, head, l, hd] = (X @ W^T + bias) split into heads.
// X: [BATCH*L, 768] f32,  W: [768, 768] f32 (torch Linear [out,in]),
// dst: [BATCH, 12, L, 64] bf16.
// Block = 256 threads (8 waves); tile 128(M) x 64(N); K-step 32.
// Each wave owns a 32x32 quadrant = 2x2 WMMA 16x16 tiles.
// ---------------------------------------------------------------------------
template <int L>
__global__ __launch_bounds__(256)
void proj_gemm_kernel(const float* __restrict__ X, const float* __restrict__ W,
                      const float* __restrict__ bias, __bf16* __restrict__ dst)
{
  __shared__ __align__(32) __bf16 As[128 * 32];  // 8 KB
  __shared__ __align__(32) __bf16 Bs[64 * 32];   // 4 KB

  const int tid  = threadIdx.x;
  const int lane = tid & 31;
  const int wave = tid >> 5;
  const int wr   = wave >> 1;            // 0..3 -> M sub-tile
  const int wc   = wave & 1;             // 0..1 -> N sub-tile
  const int m0   = blockIdx.x * 128;
  const int n0   = blockIdx.y * 64;
  const int lcol = lane & 15;
  const int hs   = (lane < 16) ? 0 : 1;  // half-wave select

  v8f acc[2][2] = {};

  for (int k0 = 0; k0 < HDIM; k0 += 32) {
    __syncthreads();
    // Stage A tile 128x32: f32 -> bf16 (1024 float4 slots, 4 per thread)
#pragma unroll
    for (int i = 0; i < 4; ++i) {
      int idx = tid + 256 * i;
      int r   = idx >> 3;
      int c4  = (idx & 7) << 2;
      const float4 v = *(const float4*)(X + (size_t)(m0 + r) * HDIM + k0 + c4);
      __bf16* p = &As[r * 32 + c4];
      p[0] = (__bf16)v.x; p[1] = (__bf16)v.y;
      p[2] = (__bf16)v.z; p[3] = (__bf16)v.w;
    }
    // Stage B tile 64x32 from W rows ([out,in] storage is B^T-ready)
#pragma unroll
    for (int i = 0; i < 2; ++i) {
      int idx = tid + 256 * i;
      int r   = idx >> 3;
      int c4  = (idx & 7) << 2;
      const float4 v = *(const float4*)(W + (size_t)(n0 + r) * HDIM + k0 + c4);
      __bf16* p = &Bs[r * 32 + c4];
      p[0] = (__bf16)v.x; p[1] = (__bf16)v.y;
      p[2] = (__bf16)v.z; p[3] = (__bf16)v.w;
    }
    __syncthreads();

    // A fragment: 16x32, lane = row (M), halves hold K runs split by half-wave
    v16bf afrag[2], bfrag[2];
#pragma unroll
    for (int mt = 0; mt < 2; ++mt) {
      int m = wr * 32 + mt * 16 + lcol;
      const __bf16* row = &As[m * 32 + hs * 8];
      afrag[mt] = cat8(*(const v8bf*)(row), *(const v8bf*)(row + 16));
    }
    // B fragment: 32x16, lane = col (N), halves hold K runs split by half-wave
#pragma unroll
    for (int nt = 0; nt < 2; ++nt) {
      int n = wc * 32 + nt * 16 + lcol;
      bfrag[nt] = cat8(*(const v8bf*)(&Bs[n * 32 + hs * 8]), *(const v8bf*)(&Bs[n * 32 + hs * 8 + 16]));
    }
#pragma unroll
    for (int mt = 0; mt < 2; ++mt)
#pragma unroll
      for (int nt = 0; nt < 2; ++nt)
        acc[mt][nt] = wmma_bf16(afrag[mt], bfrag[nt], acc[mt][nt]);
  }

  // Epilogue: bias add, scatter into [BATCH, 12, L, 64] bf16 head layout.
#pragma unroll
  for (int mt = 0; mt < 2; ++mt) {
#pragma unroll
    for (int nt = 0; nt < 2; ++nt) {
      int ncol  = n0 + wc * 32 + nt * 16 + lcol;
      int head  = ncol >> 6;
      int hd    = ncol & 63;
      float bv  = bias[ncol];
#pragma unroll
      for (int r = 0; r < 8; ++r) {
        int mrow = m0 + wr * 32 + mt * 16 + r + hs * 8;
        int b    = mrow / L;            // L is a power-of-two constant -> shift
        int lq   = mrow - b * L;
        dst[(((size_t)b * NH + head) * L + lq) * HD + hd] =
            (__bf16)(acc[mt][nt][r] + bv);
      }
    }
  }
}

// ---------------------------------------------------------------------------
// Flash attention over 2304 keys (32 word chunks + 4 entity chunks of 64).
// Word-key chunks score with Qa/Kw; entity-key chunks with Qb/Ke.
// Block = 128 threads (4 waves); each wave owns 16 query rows.
// K tile is fetched by the Tensor Data Mover (wave 0, TENSORcnt-tracked);
// V tile is staged transposed by the block for contiguous P*V B-fragments.
// ---------------------------------------------------------------------------
template <int LQ>
__global__ __launch_bounds__(128)
void flash_attn_kernel(const __bf16* __restrict__ Qa, const __bf16* __restrict__ Qb,
                       const __bf16* __restrict__ Kw, const __bf16* __restrict__ Ke,
                       const __bf16* __restrict__ Vw, const __bf16* __restrict__ Ve,
                       const float* __restrict__ mask, float* __restrict__ out)
{
  __shared__ __align__(32) __bf16 Ks[64 * 64];       // keys row-major     8 KB
  __shared__ __align__(32) __bf16 Vt[64 * 64];       // values [hd][key]   8 KB
  __shared__ __align__(32) __bf16 Ps[4][16 * 64];    // per-wave P tiles   8 KB

  const int tid  = threadIdx.x;
  const int lane = tid & 31;
  const int wave = tid >> 5;
  const int lcol = lane & 15;
  const int hs   = (lane < 16) ? 0 : 1;
  const int bh   = blockIdx.y;           // b*12 + h
  const int b    = bh / NH;
  const int h    = bh - b * NH;
  const int q0   = blockIdx.x * 64 + wave * 16;

  // Preload both Q variants as A-fragments (2 K-chunks of 32 each).
  v16bf qa[2], qb[2];
  {
    const __bf16* QaB = Qa + (size_t)bh * LQ * HD;
    const __bf16* QbB = Qb + (size_t)bh * LQ * HD;
#pragma unroll
    for (int kc = 0; kc < 2; ++kc) {
      const __bf16* ra = QaB + (size_t)(q0 + lcol) * HD + kc * 32 + hs * 8;
      const __bf16* rb = QbB + (size_t)(q0 + lcol) * HD + kc * 32 + hs * 8;
      qa[kc] = cat8(*(const v8bf*)(ra), *(const v8bf*)(ra + 16));
      qb[kc] = cat8(*(const v8bf*)(rb), *(const v8bf*)(rb + 16));
    }
  }

  v8f o_acc[4] = {};
  float m_i[8], l_i[8];
#pragma unroll
  for (int r = 0; r < 8; ++r) { m_i[r] = -1e30f; l_i[r] = 0.f; }

  const float scale = 0.125f;                   // 1/sqrt(64)
  const float* maskB = mask + (size_t)b * LSUM;
  const unsigned ks_lds = lds_offset(&Ks[0]);
  const int NCHUNK = LSUM / 64;                 // 36
  const int WORDCH = LW / 64;                   // 32

  for (int ch = 0; ch < NCHUNK; ++ch) {
    const bool isEnt = (ch >= WORDCH);
    const __bf16* Ksrc = isEnt
        ? Ke + ((size_t)bh * LE + (ch - WORDCH) * 64) * HD
        : Kw + ((size_t)bh * LW + ch * 64) * HD;
    const __bf16* Vsrc = isEnt
        ? Ve + ((size_t)bh * LE + (ch - WORDCH) * 64) * HD
        : Vw + ((size_t)bh * LW + ch * 64) * HD;

    __syncthreads();   // prior iteration's Ks/Vt consumers are done

    // K tile: async DMA via the Tensor Data Mover (wave 0 only; EXEC ignored).
    if (wave == 0) {
      tdm_load_tile_bf16(ks_lds, Ksrc, /*rows=*/64, /*cols=*/64);
    }
    // V tile: staged transposed ([hd][key]) so P*V B-fragments are contiguous.
#pragma unroll
    for (int i = 0; i < 4; ++i) {
      int idx = tid + 128 * i;
      int r   = idx >> 3;
      int c8  = (idx & 7) << 3;
      v8bf vv = *(const v8bf*)(Vsrc + (size_t)r * HD + c8);
#pragma unroll
      for (int e = 0; e < 8; ++e) Vt[(c8 + e) * 64 + r] = vv[e];
    }
    if (wave == 0) {
      __builtin_amdgcn_s_wait_tensorcnt(0);     // drain TDM before the barrier
    }
    __syncthreads();

    // S = Q * K^T  (4 key sub-tiles of 16, 2 K-dim steps of 32)
    v8f s[4] = {};
    const v16bf* qsel = isEnt ? qb : qa;
#pragma unroll
    for (int kc = 0; kc < 2; ++kc) {
#pragma unroll
      for (int c = 0; c < 4; ++c) {
        int n = c * 16 + lcol;
        v16bf bf = cat8(*(const v8bf*)(&Ks[n * 64 + kc * 32 + hs * 8]), *(const v8bf*)(&Ks[n * 64 + kc * 32 + hs * 8 + 16]));
        s[c] = wmma_bf16(qsel[kc], bf, s[c]);
      }
    }

    // Scale + additive mask (per key column).
    const int keybase = ch * 64;
#pragma unroll
    for (int c = 0; c < 4; ++c) {
      float mv = maskB[keybase + c * 16 + lcol];
#pragma unroll
      for (int r = 0; r < 8; ++r) s[c][r] = s[c][r] * scale + mv;
    }

    // Online softmax: rows live in 16-lane halves of the C layout.
    float alpha[8];
#pragma unroll
    for (int r = 0; r < 8; ++r) {
      float mx = fmaxf(fmaxf(s[0][r], s[1][r]), fmaxf(s[2][r], s[3][r]));
#pragma unroll
      for (int d = 1; d < 16; d <<= 1) mx = fmaxf(mx, __shfl_xor(mx, d, 32));
      float mnew = fmaxf(m_i[r], mx);
      float a    = __expf(m_i[r] - mnew);
      float rs   = 0.f;
#pragma unroll
      for (int c = 0; c < 4; ++c) {
        float p = __expf(s[c][r] - mnew);
        s[c][r] = p;
        rs += p;
      }
#pragma unroll
      for (int d = 1; d < 16; d <<= 1) rs += __shfl_xor(rs, d, 32);
      m_i[r]   = mnew;
      l_i[r]   = l_i[r] * a + rs;
      alpha[r] = a;
    }
#pragma unroll
    for (int ht = 0; ht < 4; ++ht)
#pragma unroll
      for (int r = 0; r < 8; ++r) o_acc[ht][r] *= alpha[r];

    // Re-layout P (C layout -> A fragment) through per-wave LDS tile.
#pragma unroll
    for (int c = 0; c < 4; ++c)
#pragma unroll
      for (int r = 0; r < 8; ++r)
        Ps[wave][(r + hs * 8) * 64 + c * 16 + lcol] = (__bf16)s[c][r];
    asm volatile("s_wait_dscnt 0" ::: "memory");   // per-wave LDS RAW fence

    // O += P * V   (key dim 64 = 2 WMMA K-steps; 4 hd tiles of 16)
#pragma unroll
    for (int c2 = 0; c2 < 2; ++c2) {
      const __bf16* prow = &Ps[wave][lcol * 64 + c2 * 32 + hs * 8];
      v16bf pf = cat8(*(const v8bf*)(prow), *(const v8bf*)(prow + 16));
#pragma unroll
      for (int ht = 0; ht < 4; ++ht) {
        v16bf vf = cat8(*(const v8bf*)(&Vt[(ht * 16 + lcol) * 64 + c2 * 32 + hs * 8]), *(const v8bf*)(&Vt[(ht * 16 + lcol) * 64 + c2 * 32 + hs * 8 + 16]));
        o_acc[ht] = wmma_bf16(pf, vf, o_acc[ht]);
      }
    }
  }

  // Epilogue: normalize by row sum, write f32 context [B, LQ, 768].
#pragma unroll
  for (int ht = 0; ht < 4; ++ht) {
    int hd = ht * 16 + lcol;
#pragma unroll
    for (int r = 0; r < 8; ++r) {
      int q = q0 + r + hs * 8;
      out[((size_t)b * LQ + q) * HDIM + h * HD + hd] = o_acc[ht][r] / l_i[r];
    }
  }
}

// ---------------------------------------------------------------------------
// Launch
// ---------------------------------------------------------------------------
extern "C" void kernel_launch(void* const* d_in, const int* in_sizes, int n_in,
                              void* d_out, int out_size, void* d_ws, size_t ws_size,
                              hipStream_t stream) {
  (void)in_sizes; (void)n_in; (void)out_size; (void)ws_size;

  const float* Xw   = (const float*)d_in[0];   // [4,2048,768]
  const float* Xe   = (const float*)d_in[1];   // [4,256,768]
  const float* mask = (const float*)d_in[2];   // [4,1,1,2304]
  auto Wp = [&](int i) { return (const float*)d_in[i]; };

  // Workspace: 12 bf16 head-major Q/K/V tensors.
  __bf16* ws = (__bf16*)d_ws;
  const size_t WP = (size_t)BATCH * NH * LW * HD;   // word-derived tensor elems
  const size_t EP = (size_t)BATCH * NH * LE * HD;   // entity-derived tensor elems
  __bf16* p[12];
  for (int i = 0; i < 6; ++i) p[i] = ws + (size_t)i * WP;
  __bf16* eb = ws + 6 * WP;
  for (int i = 0; i < 6; ++i) p[6 + i] = eb + (size_t)i * EP;
  // p[0]=Qw_w2w p[1]=Qw_w2e p[2]=Kw_w2w p[3]=Kw_e2w p[4]=Vw_word p[5]=Vw_ent
  // p[6]=Qe_e2w p[7]=Qe_e2e p[8]=Ke_w2e p[9]=Ke_e2e p[10]=Ve_word p[11]=Ve_ent

  dim3 blk(256);
  dim3 gw((BATCH * LW) / 128, HDIM / 64);   // 64 x 12
  dim3 ge((BATCH * LE) / 128, HDIM / 64);   //  8 x 12

  // Word-input projections
  proj_gemm_kernel<LW><<<gw, blk, 0, stream>>>(Xw, Wp(3),  Wp(4),  p[0]); // q
  proj_gemm_kernel<LW><<<gw, blk, 0, stream>>>(Xw, Wp(9),  Wp(10), p[1]); // w2e_q
  proj_gemm_kernel<LW><<<gw, blk, 0, stream>>>(Xw, Wp(5),  Wp(6),  p[2]); // k
  proj_gemm_kernel<LW><<<gw, blk, 0, stream>>>(Xw, Wp(17), Wp(18), p[3]); // e2w_k
  proj_gemm_kernel<LW><<<gw, blk, 0, stream>>>(Xw, Wp(7),  Wp(8),  p[4]); // v
  proj_gemm_kernel<LW><<<gw, blk, 0, stream>>>(Xw, Wp(19), Wp(20), p[5]); // e2w_v
  // Entity-input projections
  proj_gemm_kernel<LE><<<ge, blk, 0, stream>>>(Xe, Wp(15), Wp(16), p[6]);  // e2w_q
  proj_gemm_kernel<LE><<<ge, blk, 0, stream>>>(Xe, Wp(21), Wp(22), p[7]);  // e2e_q
  proj_gemm_kernel<LE><<<ge, blk, 0, stream>>>(Xe, Wp(11), Wp(12), p[8]);  // w2e_k
  proj_gemm_kernel<LE><<<ge, blk, 0, stream>>>(Xe, Wp(23), Wp(24), p[9]);  // e2e_k
  proj_gemm_kernel<LE><<<ge, blk, 0, stream>>>(Xe, Wp(13), Wp(14), p[10]); // w2e_v
  proj_gemm_kernel<LE><<<ge, blk, 0, stream>>>(Xe, Wp(25), Wp(26), p[11]); // e2e_v

  float* out_word = (float*)d_out;
  float* out_ent  = out_word + (size_t)BATCH * LW * HDIM;

  flash_attn_kernel<LW><<<dim3(LW / 64, BATCH * NH), dim3(128), 0, stream>>>(
      p[0], p[1], p[2], p[8], p[4], p[10], mask, out_word);
  flash_attn_kernel<LE><<<dim3(LE / 64, BATCH * NH), dim3(128), 0, stream>>>(
      p[6], p[7], p[3], p[9], p[5], p[11], mask, out_ent);
}
